// Head_5317169512488
// MI455X (gfx1250) — matrix operand order repaired and checked by reading, not compiled, not run on hardware
//
#include <hip/hip_runtime.h>
#include <hip/hip_bf16.h>

typedef __attribute__((ext_vector_type(8)))  _Float16 v8h;
typedef __attribute__((ext_vector_type(16))) _Float16 v16h;
typedef __attribute__((ext_vector_type(8)))  float    v8f;

#define N_EMBED 384
#define HEAD    64
#define TSEQ    256
#define BATCH   512

// ---------------------------------------------------------------------------
// helpers
// ---------------------------------------------------------------------------
// Load a 16-half B-fragment slice that is contiguous in memory (packed K = 16*(i>>3) + 8*half + (i&7)).
__device__ inline v16h ld_v16h(const _Float16* p) {
    v8h lo = *(const v8h*)(p);
    v8h hi = *(const v8h*)(p + 8);
    v16h r;
#pragma unroll
    for (int i = 0; i < 8; ++i) { r[i] = lo[i]; r[i + 8] = hi[i]; }
    return r;
}

// Load an A-fragment row slice: halves i -> K = 16*(i>>3) + 8*half + (i&7),
// i.e. two contiguous 8-half chunks at rowp + half*8 and rowp + 16 + half*8.
__device__ inline v16h ld_afrag(const _Float16* rowp, int half) {
    v8h lo = *(const v8h*)(rowp + half * 8);
    v8h hi = *(const v8h*)(rowp + 16 + half * 8);
    v16h r;
#pragma unroll
    for (int i = 0; i < 8; ++i) { r[i] = lo[i]; r[i + 8] = hi[i]; }
    return r;
}

// CDNA5 async global->LDS copy (16 bytes per lane), tracked by ASYNCcnt.
__device__ inline void async_ld16(unsigned int lds_off, const void* gptr) {
    asm volatile("global_load_async_to_lds_b128 %0, %1, off"
                 :: "v"(lds_off), "v"((unsigned long long)(size_t)gptr)
                 : "memory");
}
__device__ inline void wait_async0() {
    asm volatile("s_wait_asynccnt 0x0" ::: "memory");
}

// ---------------------------------------------------------------------------
// Kernel 0: pack Wq/Wk/Wv (f32, [C,H]) into f16 B-fragment order so the
// projection kernel reads each fragment as two contiguous b128 loads.
// pack[(m*12+chunk)*4 + ntile][lane][i] = W[(chunk*32 + 16*(i>>3) + 8*half + (i&7))*H + 16*nt + l16]
// ---------------------------------------------------------------------------
__global__ __launch_bounds__(32) void pack_w_kernel(
    const float* __restrict__ Wq, const float* __restrict__ Wk,
    const float* __restrict__ Wv, _Float16* __restrict__ packW)
{
    const int tile = blockIdx.x;            // 0..143 = (m*12+chunk)*4 + nt
    const int m    = tile / 48;
    const int rem  = tile % 48;
    const int ch   = rem >> 2;
    const int nt   = rem & 3;
    const int lane = threadIdx.x & 31;
    const int half = lane >> 4;
    const int l16  = lane & 15;

    const float* W = (m == 0) ? Wq : (m == 1) ? Wk : Wv;
    _Float16* dst = packW + ((size_t)tile * 32 + lane) * 16;
#pragma unroll
    for (int i = 0; i < 16; ++i)
        dst[i] = (_Float16)W[(size_t)(ch * 32 + (i >> 3) * 16 + half * 8 + (i & 7)) * HEAD + nt * 16 + l16];
}

// ---------------------------------------------------------------------------
// Kernel 1: QKV projection.  One block = one 16-row tile of flattened B*T.
// 4 waves; wave w owns H-columns [16w,16w+16).  12 k-steps x 3 WMMA.
// x read as float4 (b128), W fragments as packed b128.  Outputs:
//   q,k : f16 row-major [B*T, H]
//   vT  : f16 transposed [B][H][T]  (makes PV B-fragments contiguous)
// ---------------------------------------------------------------------------
__global__ __launch_bounds__(128) void qkv_proj_kernel(
    const float* __restrict__ x, const _Float16* __restrict__ packW,
    _Float16* __restrict__ qo, _Float16* __restrict__ ko,
    _Float16* __restrict__ vTo)
{
    const int row0 = blockIdx.x * 16;
    const int wave = threadIdx.x >> 5;      // n-tile
    const int lane = threadIdx.x & 31;
    const int half = lane >> 4;
    const int l16  = lane & 15;
    const int n0   = wave * 16;

    v8f acc[3] = {};

    const float4* xr = (const float4*)(x + (size_t)(row0 + l16) * N_EMBED);

    for (int ch = 0; ch < 12; ++ch) {
        const int k0 = ch * 32;
        // A fragment: 4 x float4 (16-B) contiguous chunks
        const float4 f0 = xr[(k0 + half * 8) >> 2];
        const float4 f1 = xr[((k0 + half * 8) >> 2) + 1];
        const float4 f2 = xr[(k0 + 16 + half * 8) >> 2];
        const float4 f3 = xr[((k0 + 16 + half * 8) >> 2) + 1];
        v16h a;
        a[0] = (_Float16)f0.x; a[1] = (_Float16)f0.y; a[2]  = (_Float16)f0.z; a[3]  = (_Float16)f0.w;
        a[4] = (_Float16)f1.x; a[5] = (_Float16)f1.y; a[6]  = (_Float16)f1.z; a[7]  = (_Float16)f1.w;
        a[8] = (_Float16)f2.x; a[9] = (_Float16)f2.y; a[10] = (_Float16)f2.z; a[11] = (_Float16)f2.w;
        a[12]= (_Float16)f3.x; a[13]= (_Float16)f3.y; a[14] = (_Float16)f3.z; a[15] = (_Float16)f3.w;

#pragma unroll
        for (int m = 0; m < 3; ++m) {
            const v16h bf = ld_v16h(
                packW + (((size_t)(m * 12 + ch) * 4 + wave) * 32 + lane) * 16);
            acc[m] = __builtin_amdgcn_wmma_f32_16x16x32_f16(
                false, a, false, bf, (short)0, acc[m], false, false);
        }
    }

    // C layout: VGPR r -> row M = 8*half + r ; col N = l16
    const int bb = row0 / TSEQ;
    const int tl = row0 % TSEQ;
#pragma unroll
    for (int r = 0; r < 8; ++r) {
        const int row = row0 + half * 8 + r;
        qo[(size_t)row * HEAD + n0 + l16] = (_Float16)acc[0][r];
        ko[(size_t)row * HEAD + n0 + l16] = (_Float16)acc[1][r];
    }
    v8h vv;
#pragma unroll
    for (int r = 0; r < 8; ++r) vv[r] = (_Float16)acc[2][r];
    *(v8h*)(vTo + ((size_t)bb * HEAD + n0 + l16) * TSEQ + tl + half * 8) = vv;
}

// ---------------------------------------------------------------------------
// Kernel 2: fused causal attention, flash style.  Block = 8 waves; wave w owns
// query rows [t0,t0+16).  Keys processed in two 128-key phases: each phase the
// whole block async-copies K (128x64) and V^T (64x128) into LDS once (shared
// by 8 waves), then waves run independent online-softmax loops (wave-uniform
// branches only -> EXEC all-ones for WMMA).  Even blocks skip phase 1.
// ---------------------------------------------------------------------------
__global__ __launch_bounds__(256) void attn_kernel(
    const _Float16* __restrict__ q, const _Float16* __restrict__ k,
    const _Float16* __restrict__ vT, float* __restrict__ out)
{
    __shared__ __align__(16) _Float16 kbuf[128 * HEAD];    // [tloc][h]   16 KB
    __shared__ __align__(16) _Float16 vbufT[HEAD * 128];   // [h][tloc]   16 KB
    __shared__ __align__(16) _Float16 pbuf[8][16 * 32];    // per-wave P   8 KB

    const int tid  = threadIdx.x;
    const int wave = tid >> 5;
    const int lane = tid & 31;
    const int half = lane >> 4;
    const int l16  = lane & 15;

    const int b  = blockIdx.x >> 1;
    const int t0 = (blockIdx.x & 1) * 128 + wave * 16;
    const size_t kb = (size_t)b * TSEQ * HEAD;   // q,k base
    const size_t vb = (size_t)b * HEAD * TSEQ;   // vT base

    // Q tile 16x64 -> two A fragments (contiguous b128 chunks)
    const _Float16* qrow = q + kb + (size_t)(t0 + l16) * HEAD;
    const v16h qa0 = ld_afrag(qrow, half);
    const v16h qa1 = ld_afrag(qrow + 32, half);

    v8f acc[4] = {};
    float mrow[8], lrow[8];
#pragma unroll
    for (int r = 0; r < 8; ++r) { mrow[r] = -3.0e38f; lrow[r] = 0.0f; }

    const float scale = 0.0510310363f;         // 384^-0.5 (ref scales by C)
    const int kend = t0 + 16;                  // causal extent for this wave
    const int nph  = (blockIdx.x & 1) ? 2 : 1; // uniform per block

    _Float16* my = pbuf[wave];

    for (int ph = 0; ph < nph; ++ph) {
        // ---- cooperative async preload of K / V^T for keys [128ph,128ph+128)
#pragma unroll
        for (int it = 0; it < 4; ++it) {
            const int c = tid + it * 256;               // 0..1023 16-B chunks
            const int tr = c >> 3, co = (c & 7) * 8;    // kbuf: 8 chunks/row
            async_ld16((unsigned int)(size_t)&kbuf[tr * HEAD + co],
                       k + kb + (size_t)(ph * 128 + tr) * HEAD + co);
            const int hr = c >> 4, to = (c & 15) * 8;   // vbufT: 16 chunks/row
            async_ld16((unsigned int)(size_t)&vbufT[hr * 128 + to],
                       vT + vb + (size_t)hr * TSEQ + ph * 128 + to);
        }
        wait_async0();
        __syncthreads();

        const int jcap  = ph * 128 + 128;
        const int jstop = (kend < jcap) ? kend : jcap;  // wave-uniform

        for (int j0 = ph * 128; j0 < jstop; j0 += 32) {
            const int jl = j0 - ph * 128;

            // ---- S = Q @ K^T : two 16x16 C tiles, B frags from LDS --------
            v8f s[2] = {};
#pragma unroll
            for (int sub = 0; sub < 2; ++sub) {
                {
                    const v16h bf = ld_afrag(
                        &kbuf[(jl + sub * 16 + l16) * HEAD], half);
                    s[sub] = __builtin_amdgcn_wmma_f32_16x16x32_f16(
                        false, qa0, false, bf, (short)0, s[sub], false, false);
                }
                {
                    const v16h bf = ld_afrag(
                        &kbuf[(jl + sub * 16 + l16) * HEAD + 32], half);
                    s[sub] = __builtin_amdgcn_wmma_f32_16x16x32_f16(
                        false, qa1, false, bf, (short)0, s[sub], false, false);
                }
            }

            // ---- mask + scale + row max (shfl within 16-lane group) -------
            float rmax[8];
#pragma unroll
            for (int r = 0; r < 8; ++r) {
                const int trow = t0 + half * 8 + r;
#pragma unroll
                for (int sub = 0; sub < 2; ++sub) {
                    const int key = j0 + sub * 16 + l16;
                    float val = s[sub][r] * scale;
                    val = (key > trow) ? -3.0e38f : val;   // select, no branch
                    s[sub][r] = val;
                }
                float mx = fmaxf(s[0][r], s[1][r]);
#pragma unroll
                for (int off = 8; off >= 1; off >>= 1)
                    mx = fmaxf(mx, __shfl_xor(mx, off, 32));
                rmax[r] = mx;
            }

            // ---- online softmax -------------------------------------------
            float alpha[8];
#pragma unroll
            for (int r = 0; r < 8; ++r) {
                const float mnew = fmaxf(mrow[r], rmax[r]);
                alpha[r] = __expf(mrow[r] - mnew);
                mrow[r]  = mnew;
                float psum = 0.0f;
#pragma unroll
                for (int sub = 0; sub < 2; ++sub) {
                    const float p = __expf(s[sub][r] - mnew);
                    s[sub][r] = p;
                    psum += p;
                }
#pragma unroll
                for (int off = 8; off >= 1; off >>= 1)
                    psum += __shfl_xor(psum, off, 32);
                lrow[r] = lrow[r] * alpha[r] + psum;
            }
#pragma unroll
            for (int nt = 0; nt < 4; ++nt)
#pragma unroll
                for (int r = 0; r < 8; ++r)
                    acc[nt][r] *= alpha[r];

            // ---- transpose P through per-wave LDS (C layout -> A layout) --
#pragma unroll
            for (int sub = 0; sub < 2; ++sub)
#pragma unroll
                for (int r = 0; r < 8; ++r)
                    my[(half * 8 + r) * 32 + sub * 16 + l16] = (_Float16)s[sub][r];

            const v16h pa = ld_afrag(&my[l16 * 32], half);  // same-wave RAW

            // ---- O += P @ V (B frags contiguous from V^T in LDS) ----------
#pragma unroll
            for (int nt = 0; nt < 4; ++nt) {
                const v16h bf = ld_afrag(
                    &vbufT[(nt * 16 + l16) * 128 + jl], half);
                acc[nt] = __builtin_amdgcn_wmma_f32_16x16x32_f16(
                    false, pa, false, bf, (short)0, acc[nt], false, false);
            }
        }
        __syncthreads();   // protect LDS before next phase's preload
    }

    // ---- normalize and store fp32 output ----------------------------------
    float inv[8];
#pragma unroll
    for (int r = 0; r < 8; ++r) inv[r] = 1.0f / lrow[r];
#pragma unroll
    for (int nt = 0; nt < 4; ++nt)
#pragma unroll
        for (int r = 0; r < 8; ++r) {
            const int trow = t0 + half * 8 + r;
            out[kb + (size_t)trow * HEAD + nt * 16 + l16] = acc[nt][r] * inv[r];
        }
}

// ---------------------------------------------------------------------------
extern "C" void kernel_launch(void* const* d_in, const int* in_sizes, int n_in,
                              void* d_out, int out_size, void* d_ws, size_t ws_size,
                              hipStream_t stream) {
    const float* x  = (const float*)d_in[0];
    const float* Wq = (const float*)d_in[1];
    const float* Wk = (const float*)d_in[2];
    const float* Wv = (const float*)d_in[3];
    float* out = (float*)d_out;

    const size_t BT = (size_t)BATCH * TSEQ;
    _Float16* qws   = (_Float16*)d_ws;        // BT*64 f16
    _Float16* kws   = qws  + BT * HEAD;       // BT*64 f16
    _Float16* vTws  = kws  + BT * HEAD;       // [B][H][T] f16
    _Float16* packW = vTws + BT * HEAD;       // 3*12*4*32*16 = 73728 f16

    pack_w_kernel<<<dim3(144), dim3(32), 0, stream>>>(Wq, Wk, Wv, packW);
    qkv_proj_kernel<<<dim3((unsigned)(BT / 16)), dim3(128), 0, stream>>>(
        x, packW, qws, kws, vTws);
    attn_kernel<<<dim3(BATCH * 2), dim3(256), 0, stream>>>(
        qws, kws, vTws, out);
}
